// MyModel_58007828300239
// MI455X (gfx1250) — compile-verified, matrix operand order checked by reading, not run on hardware
//
#include <hip/hip_runtime.h>
#include <hip/hip_bf16.h>
#include <stdint.h>

#define VOCAB 4000
#define HID   1024
#define G3    3072
#define BATCH 64
#define SEQ   256

typedef __attribute__((ext_vector_type(16))) __bf16        v16bf;
typedef __attribute__((ext_vector_type(8)))  float         v8f;
typedef __attribute__((ext_vector_type(4)))  unsigned int  v4u;
typedef __attribute__((ext_vector_type(8)))  int           v8i;
typedef __attribute__((ext_vector_type(4)))  int           v4i;

__device__ __forceinline__ unsigned short f32_to_bf16_bits(float f) {
  unsigned int u = __float_as_uint(f);
  u += 0x7FFFu + ((u >> 16) & 1u);   // round-to-nearest-even
  return (unsigned short)(u >> 16);
}

__device__ __forceinline__ float fast_rcp(float x) {
#if __has_builtin(__builtin_amdgcn_rcpf)
  return __builtin_amdgcn_rcpf(x);       // v_rcp_f32
#else
  return 1.0f / x;
#endif
}

__device__ __forceinline__ float fast_sigmoid(float x) {
  return fast_rcp(1.0f + __expf(-x));    // v_exp_f32 + v_rcp_f32
}

__device__ __forceinline__ float fast_tanh(float x) {
#if __has_builtin(__builtin_amdgcn_tanhf)
  return __builtin_amdgcn_tanhf(x);      // gfx1250 v_tanh_f32
#else
  float e = __expf(2.0f * x);
  return 1.0f - 2.0f * fast_rcp(e + 1.0f);
#endif
}

// ---------------------------------------------------------------------------
// Tensor Data Mover: 2D tile (tile_k x tile_rows) of bf16, row stride in elems.
// D# layout per CDNA5 ISA. Zero-fills reads past
// tensor_dim1 (used for the ragged N edge of the output GEMM).
// ---------------------------------------------------------------------------
__device__ __forceinline__ void tdm_load_2d_bf16(
    unsigned int lds_off, const void* gptr,
    unsigned int tile_k, unsigned int tile_rows,
    unsigned int tensor_rows, unsigned long long row_stride_elems)
{
  unsigned long long ga = (unsigned long long)(uintptr_t)gptr;
  v4u g0;
  g0.x = 1u;                                            // count=1, user mode
  g0.y = lds_off;                                       // lds_addr [63:32]
  g0.z = (unsigned int)(ga & 0xFFFFFFFFu);              // global_addr[31:0]
  g0.w = (unsigned int)((ga >> 32) & 0x01FFFFFFu)       // global_addr[56:32]
       | (2u << 30);                                    // type=2 ("image")
  unsigned int dim0 = tile_k;                           // tensor_dim0 (K extent)
  unsigned int dim1 = tensor_rows;                      // tensor_dim1 (rows, OOB clamp)
  v8i g1;
  g1[0] = (int)(1u << 16);                                          // data_size=1 (2B)
  g1[1] = (int)((dim0 & 0xFFFFu) << 16);                            // tensor_dim0[15:0]
  g1[2] = (int)(((dim0 >> 16) & 0xFFFFu) | ((dim1 & 0xFFFFu) << 16));
  g1[3] = (int)(((dim1 >> 16) & 0xFFFFu) | ((tile_k & 0xFFFFu) << 16)); // tile_dim0
  g1[4] = (int)(tile_rows & 0xFFFFu);                               // tile_dim1; tile_dim2=0
  g1[5] = (int)(unsigned int)(row_stride_elems & 0xFFFFFFFFull);    // dim0_stride[31:0]
  g1[6] = (int)(unsigned int)((row_stride_elems >> 32) & 0xFFFFull);// dim0_stride[47:32]
  g1[7] = 0;
  v4i z4 = {0, 0, 0, 0};
#if __clang_major__ >= 23
  v8i z8 = {0, 0, 0, 0, 0, 0, 0, 0};
  __builtin_amdgcn_tensor_load_to_lds(g0, g1, z4, z4, z8, 0);
#else
  __builtin_amdgcn_tensor_load_to_lds(g0, g1, z4, z4, 0);
#endif
}

// ---------------------------------------------------------------------------
// f32 -> bf16 (raw u16) convert
// ---------------------------------------------------------------------------
__global__ void f32_to_bf16_kernel(const float* __restrict__ src,
                                   unsigned short* __restrict__ dst, int n) {
  int i = blockIdx.x * blockDim.x + threadIdx.x;
  if (i < n) dst[i] = f32_to_bf16_bits(src[i]);
}

// ---------------------------------------------------------------------------
// 32x32 LDS-tiled transpose: src (rows x cols) f32 -> dst (cols x rows)
// ---------------------------------------------------------------------------
__global__ void __launch_bounds__(256) transpose_f32_kernel(
    const float* __restrict__ src, float* __restrict__ dst, int rows, int cols) {
  __shared__ float tile[32][33];
  const int tx = threadIdx.x & 31;
  const int ty = threadIdx.x >> 5;           // 0..7
  const int bx = blockIdx.x * 32;            // col block
  const int by = blockIdx.y * 32;            // row block
  #pragma unroll
  for (int j = 0; j < 32; j += 8) {
    int r = by + ty + j, c = bx + tx;
    tile[ty + j][tx] = (r < rows && c < cols) ? src[(size_t)r * cols + c] : 0.0f;
  }
  __syncthreads();
  #pragma unroll
  for (int j = 0; j < 32; j += 8) {
    int r = bx + ty + j, c = by + tx;        // dst is (cols x rows)
    if (r < cols && c < rows) dst[(size_t)r * rows + c] = tile[tx][ty + j];
  }
}

// ---------------------------------------------------------------------------
// x_proj[p, :] = W_ihT[token(p), :] + b_ih ;  p = t*64 + b, token = inputs[b, t]
// ---------------------------------------------------------------------------
__global__ void __launch_bounds__(256) xproj_kernel(
    const int* __restrict__ tokens,      // (64, 256) int32
    const float* __restrict__ WihT,      // (4000, 3072)
    const float* __restrict__ b_ih,      // (3072)
    float* __restrict__ xproj)           // (16384, 3072)
{
  const int p = blockIdx.x;
  const int t = p >> 6, b = p & 63;
  const int tok = tokens[b * SEQ + t];
  const float4* src = reinterpret_cast<const float4*>(WihT + (size_t)tok * G3);
  const float4* bi  = reinterpret_cast<const float4*>(b_ih);
  float4* dst       = reinterpret_cast<float4*>(xproj + (size_t)p * G3);
  for (int j = threadIdx.x; j < G3 / 4; j += 256) {
    float4 v = src[j], bb = bi[j];
    v.x += bb.x; v.y += bb.y; v.z += bb.z; v.w += bb.w;
    dst[j] = v;
  }
}

// ---------------------------------------------------------------------------
// One GRU step, fully fused: gh = h_bf16 @ W_hh^T (+b_hh), gates, h update.
// grid = 16 WGs x 128 thr (4 waves). Wave handles 16 hidden cols across all
// 3 gates; 12 v_wmma_f32_16x16x32_bf16 per K-iteration (M=64 -> 4 M-tiles).
// ---------------------------------------------------------------------------
__global__ void __launch_bounds__(128) gru_step_kernel(
    const unsigned short* __restrict__ Aprev,   // (64,1024) bf16: h_{t-1}
    const unsigned short* __restrict__ Whh,     // (3072,1024) bf16
    const float* __restrict__ xproj_t,          // (64, 3072) f32
    const float* __restrict__ b_hh,             // (3072)
    const float* __restrict__ h_old,            // (64,1024) f32
    float* __restrict__ h_new,                  // (64,1024) f32
    unsigned short* __restrict__ h_out)         // (64,1024) bf16 (gru_out[t])
{
  const int wave = threadIdx.x >> 5, lane = threadIdx.x & 31;
  const int ln = lane & 15, lh = lane >> 4;
  const int c0 = blockIdx.x * 64 + wave * 16;   // hidden column base

  v8f acc[3][4];
  #pragma unroll
  for (int g = 0; g < 3; ++g)
    #pragma unroll
    for (int mt = 0; mt < 4; ++mt)
      acc[g][mt] = (v8f){0.f, 0.f, 0.f, 0.f, 0.f, 0.f, 0.f, 0.f};

  for (int k0 = 0; k0 < HID; k0 += 32) {
    v16bf a[4];
    #pragma unroll
    for (int mt = 0; mt < 4; ++mt)
      a[mt] = *reinterpret_cast<const v16bf*>(
          Aprev + (size_t)(mt * 16 + ln) * HID + k0 + lh * 16);
    #pragma unroll
    for (int g = 0; g < 3; ++g) {
      v16bf b = *reinterpret_cast<const v16bf*>(
          Whh + (size_t)(g * HID + c0 + ln) * HID + k0 + lh * 16);
      #pragma unroll
      for (int mt = 0; mt < 4; ++mt)
        acc[g][mt] = __builtin_amdgcn_wmma_f32_16x16x32_bf16(
            false, a[mt], false, b, (short)0, acc[g][mt], false, false);
    }
  }

  const int col = c0 + ln;
  const float bhr = b_hh[col];
  const float bhz = b_hh[HID + col];
  const float bhn = b_hh[2 * HID + col];
  #pragma unroll
  for (int mt = 0; mt < 4; ++mt) {
    #pragma unroll
    for (int i = 0; i < 8; ++i) {
      const int m = mt * 16 + lh * 8 + i;       // C layout: VGPRi -> M=i (+8 hi half)
      const float* xrow = xproj_t + (size_t)m * G3;
      const float r = fast_sigmoid(xrow[col]           + acc[0][mt][i] + bhr);
      const float z = fast_sigmoid(xrow[HID + col]     + acc[1][mt][i] + bhz);
      const float n = fast_tanh   (xrow[2 * HID + col] + r * (acc[2][mt][i] + bhn));
      const float ho = h_old[(size_t)m * HID + col];
      const float hv = (1.0f - z) * n + z * ho;
      h_new[(size_t)m * HID + col] = hv;
      h_out[(size_t)m * HID + col] = f32_to_bf16_bits(hv);
    }
  }
}

// ---------------------------------------------------------------------------
// logits = gru_out(bf16) @ W_out^T(bf16) + b_out, f32 accumulate.
// WG = 256 thr (8 waves), macrotile 128x256, wave tile 64x64, K step 32.
// A/B tiles double-buffered in LDS via the Tensor Data Mover: wave 0 issues
// the next slice's TDM loads while all 8 waves run 16 WMMAs on the current
// slice; one barrier per K-iteration (barrier at iter i proves buf[i^1] is
// no longer being read before TDM overwrites it).
// ---------------------------------------------------------------------------
__global__ void __launch_bounds__(256) out_gemm_kernel(
    const unsigned short* __restrict__ Agru,   // (16384, 1024) bf16
    const unsigned short* __restrict__ Wout,   // (4000, 1024) bf16
    const float* __restrict__ b_out,           // (4000)
    float* __restrict__ out)                   // (16384, 4000) f32
{
  __shared__ alignas(32) unsigned short sA[2][128 * 32];   // 2 x 8 KB
  __shared__ alignas(32) unsigned short sB[2][256 * 32];   // 2 x 16 KB
  const int tid  = threadIdx.x;
  const int wave = tid >> 5, lane = tid & 31;
  const int ln = lane & 15, lh = lane >> 4;
  const int m0 = blockIdx.y * 128;
  const int n0 = blockIdx.x * 256;
  const int wm = (wave & 1) * 64;              // 2x4 wave grid, wave tile 64x64
  const int wn = (wave >> 1) * 64;

  v8f acc[4][4];
  #pragma unroll
  for (int mt = 0; mt < 4; ++mt)
    #pragma unroll
    for (int nt = 0; nt < 4; ++nt)
      acc[mt][nt] = (v8f){0.f, 0.f, 0.f, 0.f, 0.f, 0.f, 0.f, 0.f};

  // low 32 bits of a generic LDS pointer == wave-relative LDS byte offset
  const unsigned int ldsA0 = (unsigned int)(uintptr_t)&sA[0][0];
  const unsigned int ldsA1 = (unsigned int)(uintptr_t)&sA[1][0];
  const unsigned int ldsB0 = (unsigned int)(uintptr_t)&sB[0][0];
  const unsigned int ldsB1 = (unsigned int)(uintptr_t)&sB[1][0];
  const unsigned int rowsB = (unsigned int)(VOCAB - n0);  // ragged edge zero-fill

  // prologue: preload K-slice 0 into buffer 0
  if (wave == 0) {
    tdm_load_2d_bf16(ldsA0, Agru + ((size_t)m0 * HID), 32u, 128u, 128u,
                     (unsigned long long)HID);
    tdm_load_2d_bf16(ldsB0, Wout + ((size_t)n0 * HID), 32u, 256u, rowsB,
                     (unsigned long long)HID);
  }

  for (int k0 = 0; k0 < HID; k0 += 32) {
    const int cur = (k0 >> 5) & 1;
    if (wave == 0) __builtin_amdgcn_s_wait_tensorcnt(0);   // buf[cur] landed
    __syncthreads();   // publish buf[cur]; proves buf[cur^1] reads are done
    if (wave == 0 && (k0 + 32) < HID) {
      const unsigned int lA = cur ? ldsA0 : ldsA1;
      const unsigned int lB = cur ? ldsB0 : ldsB1;
      tdm_load_2d_bf16(lA, Agru + ((size_t)m0 * HID + k0 + 32), 32u, 128u, 128u,
                       (unsigned long long)HID);
      tdm_load_2d_bf16(lB, Wout + ((size_t)n0 * HID + k0 + 32), 32u, 256u, rowsB,
                       (unsigned long long)HID);
    }

    v16bf a[4], b[4];
    #pragma unroll
    for (int mt = 0; mt < 4; ++mt)
      a[mt] = *reinterpret_cast<const v16bf*>(
          &sA[cur][(wm + mt * 16 + ln) * 32 + lh * 16]);
    #pragma unroll
    for (int nt = 0; nt < 4; ++nt)
      b[nt] = *reinterpret_cast<const v16bf*>(
          &sB[cur][(wn + nt * 16 + ln) * 32 + lh * 16]);
    #pragma unroll
    for (int mt = 0; mt < 4; ++mt)
      #pragma unroll
      for (int nt = 0; nt < 4; ++nt)
        acc[mt][nt] = __builtin_amdgcn_wmma_f32_16x16x32_bf16(
            false, a[mt], false, b[nt], (short)0, acc[mt][nt], false, false);
  }

  #pragma unroll
  for (int mt = 0; mt < 4; ++mt) {
    #pragma unroll
    for (int nt = 0; nt < 4; ++nt) {
      const int v = n0 + wn + nt * 16 + ln;
      if (v < VOCAB) {
        const float bo = b_out[v];
        #pragma unroll
        for (int i = 0; i < 8; ++i) {
          const int m = m0 + wm + mt * 16 + lh * 8 + i;
          out[(size_t)m * VOCAB + v] = acc[mt][nt][i] + bo;
        }
      }
    }
  }
}

// ---------------------------------------------------------------------------
// Host-side orchestration
// ---------------------------------------------------------------------------
extern "C" void kernel_launch(void* const* d_in, const int* in_sizes, int n_in,
                              void* d_out, int out_size, void* d_ws, size_t ws_size,
                              hipStream_t stream) {
  (void)in_sizes; (void)n_in; (void)out_size; (void)ws_size;
  const int*   tokens = (const int*)d_in[0];     // (64,256)
  const float* h0     = (const float*)d_in[1];   // (1,64,1024)
  const float* W_ih   = (const float*)d_in[2];   // (3072,4000)
  const float* W_hh   = (const float*)d_in[3];   // (3072,1024)
  const float* b_ih   = (const float*)d_in[4];   // (3072)
  const float* b_hh   = (const float*)d_in[5];   // (3072)
  const float* W_out  = (const float*)d_in[6];   // (4000,1024)
  const float* b_out  = (const float*)d_in[7];   // (4000)
  float* out = (float*)d_out;

  // workspace layout (bytes, 256-aligned)
  char* ws = (char*)d_ws;
  unsigned short* whh_bf  = (unsigned short*)(ws + 0);          //  6,291,456
  unsigned short* wout_bf = (unsigned short*)(ws + 6291456);    //  8,192,000
  float*          wihT    = (float*)(ws + 14483456);            // 49,152,000
  float*          xproj   = (float*)(ws + 63635456);            // 201,326,592
  unsigned short* gruout  = (unsigned short*)(ws + 264962048);  // 33,554,432
  float*          hf32    = (float*)(ws + 298516480);           // 2 x 256 KB ping-pong
  unsigned short* hbf0    = (unsigned short*)(ws + 299040768);  // 128 KB

  // weight precision conversion + h0 init
  f32_to_bf16_kernel<<<(G3 * HID + 255) / 256, 256, 0, stream>>>(W_hh, whh_bf, G3 * HID);
  f32_to_bf16_kernel<<<(VOCAB * HID + 255) / 256, 256, 0, stream>>>(W_out, wout_bf, VOCAB * HID);
  f32_to_bf16_kernel<<<(BATCH * HID + 255) / 256, 256, 0, stream>>>(h0, hbf0, BATCH * HID);
  hipMemcpyAsync(hf32, h0, (size_t)BATCH * HID * sizeof(float),
                 hipMemcpyDeviceToDevice, stream);

  // W_ih (3072 x 4000) -> W_ihT (4000 x 3072) so the token gather is coalesced
  dim3 tg((VOCAB + 31) / 32, (G3 + 31) / 32);
  transpose_f32_kernel<<<tg, 256, 0, stream>>>(W_ih, wihT, G3, VOCAB);

  // x_proj = W_ihT[tokens] + b_ih
  xproj_kernel<<<SEQ * BATCH, 256, 0, stream>>>(tokens, wihT, b_ih, xproj);

  // sequential GRU scan: 256 fused step kernels
  for (int t = 0; t < SEQ; ++t) {
    const unsigned short* Aprev =
        (t == 0) ? hbf0 : (gruout + (size_t)(t - 1) * BATCH * HID);
    const float* holdp = hf32 + (size_t)(t & 1) * BATCH * HID;
    float*       hnewp = hf32 + (size_t)((t + 1) & 1) * BATCH * HID;
    gru_step_kernel<<<16, 128, 0, stream>>>(
        Aprev, whh_bf, xproj + (size_t)t * BATCH * G3, b_hh,
        holdp, hnewp, gruout + (size_t)t * BATCH * HID);
  }

  // logits = gru_out @ W_out^T + b_out
  dim3 og((VOCAB + 255) / 256, (SEQ * BATCH) / 128);
  out_gemm_kernel<<<og, 256, 0, stream>>>(gruout, wout_bf, b_out, out);

  // hn tail: h after t=255 lives in hf32[(255+1)&1] == hf32[0]
  hipMemcpyAsync(out + (size_t)SEQ * BATCH * VOCAB, hf32,
                 (size_t)BATCH * HID * sizeof(float),
                 hipMemcpyDeviceToDevice, stream);
}
